// SelfAttention_79852031967863
// MI455X (gfx1250) — matrix operand order repaired and checked by reading, not compiled, not run on hardware
//
#include <hip/hip_runtime.h>

// MI455X / gfx1250 fused self-attention (column-softmax variant).
// B=4, C=256, H=W=64 -> N=4096, QP=32.
// out[b,c,j] = gamma * sum_i softmax_i(q_i . k_j) * v[c,i] + x[b,c,j]
// == flash attention with Q<-k (d=32), K<-q (d=32), V<-v^T (d=256).
//
// Kernel 1 (qkv): bf16 projections via v_wmma_f32_16x16x32_bf16.
// Kernel 2 (attn): flash attention; each wave owns 16 j's and all 256 output
//   channels (16 fp32 acc tiles, big ILP); V tiles double-buffered into LDS
//   with global_load_async_to_lds_b128 (ASYNCcnt) and shared by 8 waves;
//   softmax computed exactly once per (j,i); one barrier per i-tile.

typedef __attribute__((ext_vector_type(16))) __bf16 v16bf;
typedef __attribute__((ext_vector_type(8)))  __bf16 v8bf;
typedef __attribute__((ext_vector_type(8)))  float  v8f;
typedef __attribute__((ext_vector_type(4)))  float  v4f;

#define NPOS 4096
#define CIN  256
#define QPD  32

static __device__ inline v16bf cat8(v8bf lo, v8bf hi) {
  v16bf r;
#pragma unroll
  for (int i = 0; i < 8; ++i) { r[i] = lo[i]; r[i + 8] = hi[i]; }
  return r;
}

// A-matrix fragment (16x32 bf16): lane half h holds K chunk [16h,16h+16)
// (contiguous order, matching the B fragments).
static __device__ inline v16bf load_a_bf16row(const __bf16* row, int h) {
  v8bf lo = *(const v8bf*)(row + 16 * h);
  v8bf hi = *(const v8bf*)(row + 16 * h + 8);
  return cat8(lo, hi);
}

static __device__ inline v16bf load_a_f32row(const float* rowk, int h) {
  const float* p0 = rowk + 16 * h;
  const float* p1 = rowk + 16 * h + 8;
  v4f a0 = *(const v4f*)(p0);
  v4f a1 = *(const v4f*)(p0 + 4);
  v4f b0 = *(const v4f*)(p1);
  v4f b1 = *(const v4f*)(p1 + 4);
  v16bf r;
#pragma unroll
  for (int e = 0; e < 4; ++e) {
    r[e]      = (__bf16)a0[e];
    r[4 + e]  = (__bf16)a1[e];
    r[8 + e]  = (__bf16)b0[e];
    r[12 + e] = (__bf16)b1[e];
  }
  return r;
}

// Generic pointers into LDS carry the byte offset in the low 32 bits
// (flat aperture translation).
static __device__ inline unsigned lds_off_of(const void* p) {
  return (unsigned)(unsigned long long)p;
}

// CDNA5 async global->LDS copy, tracked by ASYNCcnt (ISA 10.x / 15.18.3).
static __device__ inline void async_ld_b128(unsigned lds_byte_off,
                                            const __bf16* gaddr) {
  asm volatile("global_load_async_to_lds_b128 %0, %1, off"
               :: "v"(lds_byte_off), "v"(gaddr)
               : "memory");
}
static __device__ inline void wait_async0() {
  asm volatile("s_wait_asynccnt 0" ::: "memory");
}

// ---------------------------------------------------------------------------
// Kernel 1: q/k/v projections. grid = B * (N/128), block = 256 (8 waves).
// Outputs: qb,kb bf16 [B,N,32]; vb bf16 [B,C,N].
// ---------------------------------------------------------------------------
__global__ __launch_bounds__(256)
void qkv_kernel(const float* __restrict__ x,
                const float* __restrict__ Wq, const float* __restrict__ bq,
                const float* __restrict__ Wk, const float* __restrict__ bk,
                const float* __restrict__ Wv, const float* __restrict__ bv,
                __bf16* __restrict__ qb, __bf16* __restrict__ kb,
                __bf16* __restrict__ vb) {
  __shared__ __align__(32) __bf16 xt[128][32];  // [n_local][k_local]

  const int b    = blockIdx.x >> 5;
  const int n0   = (blockIdx.x & 31) * 128;
  const int tid  = threadIdx.x;
  const int lane = tid & 31;
  const int wave = tid >> 5;
  const int ln   = lane & 15;
  const int lh   = lane >> 4;

  const float* xb = x + (size_t)b * CIN * NPOS;

  v16bf Bfrag[8];

#pragma unroll
  for (int kk = 0; kk < 8; ++kk) {
    const int k0  = kk * 32;
    const int row = tid >> 3;
    const int col = (tid & 7) * 16;
    const float* src = xb + (size_t)(k0 + row) * NPOS + n0 + col;
#pragma unroll
    for (int u = 0; u < 4; ++u) {
      v4f v = *(const v4f*)(src + 4 * u);
#pragma unroll
      for (int e = 0; e < 4; ++e)
        xt[col + 4 * u + e][row] = (__bf16)v[e];
    }
    if (kk < 7) __builtin_prefetch(src + 32 * NPOS, 0, 1);
    __syncthreads();
    Bfrag[kk] = *(const v16bf*)(&xt[wave * 16 + ln][16 * lh]);
    __syncthreads();
  }

  const int nw = n0 + wave * 16;

#pragma unroll 1
  for (int mt = 0; mt < 20; ++mt) {
    const float* W;
    const float* bias;
    int mrow, kind;
    if (mt < 2)      { W = Wq; bias = bq; mrow = mt * 16;       kind = 0; }
    else if (mt < 4) { W = Wk; bias = bk; mrow = (mt - 2) * 16; kind = 1; }
    else             { W = Wv; bias = bv; mrow = (mt - 4) * 16; kind = 2; }

    v8f acc = {};
#pragma unroll
    for (int kk = 0; kk < 8; ++kk) {
      const float* wrow = W + (size_t)(mrow + ln) * CIN + kk * 32;
      v16bf A = load_a_f32row(wrow, lh);
      acc = __builtin_amdgcn_wmma_f32_16x16x32_bf16(
          false, A, false, Bfrag[kk], (short)0, acc, false, false);
    }
#pragma unroll
    for (int r = 0; r < 8; ++r) {
      const int m   = mrow + r + 8 * lh;
      const int n   = nw + ln;
      const float v = acc[r] + bias[m];
      if (kind == 0)
        qb[(size_t)b * NPOS * QPD + (size_t)n * QPD + m] = (__bf16)v;
      else if (kind == 1)
        kb[(size_t)b * NPOS * QPD + (size_t)n * QPD + m] = (__bf16)v;
      else
        vb[(size_t)b * CIN * NPOS + (size_t)m * NPOS + n] = (__bf16)v;
    }
  }
}

// ---------------------------------------------------------------------------
// Kernel 2: flash attention. grid = B * (N/128) = 128 blocks, 256 threads.
// Wave w owns j in [jb + 16w, jb + 16w + 16) and ALL 256 channels.
// V tile [32i x 256c] double-buffered in LDS via async loads, shared by all
// waves; softmax per wave (16-lane half-group shuffles); no redundant work.
// ---------------------------------------------------------------------------
__global__ __launch_bounds__(256)
void attn_kernel(const float* __restrict__ x, const float* __restrict__ gamma_p,
                 const __bf16* __restrict__ qb, const __bf16* __restrict__ kb,
                 const __bf16* __restrict__ vb, float* __restrict__ out) {
  __shared__ __align__(32) __bf16 Vl[2][256][32];  // 32 KB: V double buffer
  __shared__ __align__(32) __bf16 Pw[8][16][32];   //  8 KB: per-wave P tile
  __shared__ __align__(16) float  Tw[8][32][16];   // 16 KB: per-wave transpose

  const int b    = blockIdx.x >> 5;
  const int jb   = (blockIdx.x & 31) * 128;
  const int tid  = threadIdx.x;
  const int lane = tid & 31;
  const int wave = tid >> 5;
  const int ln   = lane & 15;
  const int lh   = lane >> 4;
  const int j0   = jb + wave * 16;

  const __bf16* qbB = qb + (size_t)b * NPOS * QPD;
  const __bf16* kbB = kb + (size_t)b * NPOS * QPD;
  const __bf16* vbB = vb + (size_t)b * CIN * NPOS;

  // Q' = k-projection rows j0..j0+15 (A-matrix of S, K = QP = 32).
  const v16bf Qf = load_a_bf16row(kbB + (size_t)(j0 + ln) * QPD, lh);

  v8f acc[16];
#pragma unroll
  for (int t = 0; t < 16; ++t) acc[t] = (v8f){};

  float mrow[8], lrow[8];
#pragma unroll
  for (int r = 0; r < 8; ++r) { mrow[r] = -3.0e38f; lrow[r] = 0.0f; }

  // Stage first V tile: thread tid loads row c=tid, 32 bf16 (64B = 4 b128).
  {
    const __bf16* src = vbB + (size_t)tid * NPOS;
    const unsigned dst = lds_off_of(&Vl[0][tid][0]);
#pragma unroll
    for (int u = 0; u < 4; ++u)
      async_ld_b128(dst + 16 * u, src + 8 * u);
    wait_async0();
    __syncthreads();
  }

  for (int i0 = 0, it = 0; i0 < NPOS; i0 += 32, ++it) {
    const int buf = it & 1;

    // Issue async copy of the NEXT V tile into the other buffer.
    if (i0 + 32 < NPOS) {
      const __bf16* src = vbB + (size_t)tid * NPOS + (i0 + 32);
      const unsigned dst = lds_off_of(&Vl[buf ^ 1][tid][0]);
#pragma unroll
      for (int u = 0; u < 4; ++u)
        async_ld_b128(dst + 16 * u, src + 8 * u);
    }

    // S = Q' x K'^T for this wave's 16 j's, 32 i's (K = QP = 32).
    v16bf B0 = *(const v16bf*)(qbB + (size_t)(i0 + ln) * QPD + 16 * lh);
    v16bf B1 = *(const v16bf*)(qbB + (size_t)(i0 + 16 + ln) * QPD + 16 * lh);
    v8f s0 = {}, s1 = {};
    s0 = __builtin_amdgcn_wmma_f32_16x16x32_bf16(false, Qf, false, B0,
                                                 (short)0, s0, false, false);
    s1 = __builtin_amdgcn_wmma_f32_16x16x32_bf16(false, Qf, false, B1,
                                                 (short)0, s1, false, false);
    __builtin_prefetch(qbB + (size_t)(i0 + 32 + ln) * QPD, 0, 1);

    // Online softmax over i. Rows j live across 16-lane half-groups:
    // VGPR r, lane l -> j = r + 8*(l/16), i = l%16 (+16 for s1).
    float al[8];
#pragma unroll
    for (int r = 0; r < 8; ++r) {
      float mx = fmaxf(s0[r], s1[r]);
#pragma unroll
      for (int off = 1; off < 16; off <<= 1)
        mx = fmaxf(mx, __shfl_xor(mx, off, 32));
      const float mnew = fmaxf(mrow[r], mx);
      al[r] = __expf(mrow[r] - mnew);
      const float p0 = __expf(s0[r] - mnew);
      const float p1 = __expf(s1[r] - mnew);
      lrow[r] = lrow[r] * al[r] + (p0 + p1);  // per-lane partial denominator
      mrow[r] = mnew;
      const int j = r + 8 * lh;
      Pw[wave][j][ln]      = (__bf16)p0;
      Pw[wave][j][ln + 16] = (__bf16)p1;
    }

    // P as A-fragment (same-wave LDS dependency; hardware DS ordering).
    v16bf Pf = load_a_bf16row(&Pw[wave][ln][0], lh);

    // Rescale accumulators (al is uniform within each 16-lane half-group).
#pragma unroll
    for (int t = 0; t < 16; ++t)
#pragma unroll
      for (int r = 0; r < 8; ++r) acc[t][r] *= al[r];

    // O += P * V' over all 16 channel tiles (K = 32 i's).
#pragma unroll
    for (int ct = 0; ct < 16; ++ct) {
      v16bf Vf = *(const v16bf*)(&Vl[buf][ct * 16 + ln][16 * lh]);
      acc[ct] = __builtin_amdgcn_wmma_f32_16x16x32_bf16(
          false, Pf, false, Vf, (short)0, acc[ct], false, false);
    }

    wait_async0();      // our async writes into Vl[buf^1] have landed
    __syncthreads();    // everyone done reading Vl[buf], writes visible
  }

  // Final denominators: reduce per-lane partials across the 16-lane group.
  float linv[8];
#pragma unroll
  for (int r = 0; r < 8; ++r) {
    float s = lrow[r];
#pragma unroll
    for (int off = 1; off < 16; off <<= 1)
      s += __shfl_xor(s, off, 32);
    linv[r] = 1.0f / s;
  }

  // Epilogue: per-wave transpose through LDS, then coalesced gamma*O + x.
  const float g = *gamma_p;
#pragma unroll 1
  for (int ch = 0; ch < 8; ++ch) {  // 2 channel tiles (32 c) per chunk
#pragma unroll
    for (int t = 0; t < 2; ++t)
#pragma unroll
      for (int r = 0; r < 8; ++r)
        Tw[wave][t * 16 + ln][r + 8 * lh] = acc[2 * ch + t][r] * linv[r];
    // Same-wave DS write->read ordering handled by hardware in-order DS.
    const size_t base =
        (size_t)b * CIN * NPOS + (size_t)(ch * 32 + lane) * NPOS + j0;
#pragma unroll
    for (int u = 0; u < 4; ++u) {
      v4f xv = *(const v4f*)(x + base + 4 * u);
      v4f ov;
#pragma unroll
      for (int e = 0; e < 4; ++e)
        ov[e] = g * Tw[wave][lane][4 * u + e] + xv[e];
      *(v4f*)(out + base + 4 * u) = ov;
    }
  }
}

// ---------------------------------------------------------------------------
extern "C" void kernel_launch(void* const* d_in, const int* in_sizes, int n_in,
                              void* d_out, int out_size, void* d_ws,
                              size_t ws_size, hipStream_t stream) {
  (void)in_sizes; (void)n_in; (void)out_size; (void)ws_size;
  const float* x  = (const float*)d_in[0];
  const float* Wq = (const float*)d_in[1];
  const float* bq = (const float*)d_in[2];
  const float* Wk = (const float*)d_in[3];
  const float* bk = (const float*)d_in[4];
  const float* Wv = (const float*)d_in[5];
  const float* bv = (const float*)d_in[6];
  const float* gm = (const float*)d_in[7];
  float* out = (float*)d_out;

  // Workspace: qb bf16 [B,N,32] @0 (1MB); kb @1MB (1MB); vb bf16 [B,C,N] @2MB
  // (8MB). 10MB total -> L2-resident during attention.
  char* ws = (char*)d_ws;
  __bf16* qb = (__bf16*)(ws);
  __bf16* kb = (__bf16*)(ws + ((size_t)1 << 20));
  __bf16* vb = (__bf16*)(ws + ((size_t)2 << 20));

  qkv_kernel<<<dim3(4 * 32), dim3(256), 0, stream>>>(x, Wq, bq, Wk, bk, Wv, bv,
                                                     qb, kb, vb);
  attn_kernel<<<dim3(4 * 32), dim3(256), 0, stream>>>(x, gm, qb, kb, vb, out);
}
